// HypergraphEmbeddingAggregator_58944131170871
// MI455X (gfx1250) — matrix operand order repaired and checked by reading, compiled, not run on hardware
//
#include <hip/hip_runtime.h>
#include <hip/hip_bf16.h>

// ---------------- problem constants ----------------
#define T_DIM   1024
#define IN_DIM  128
#define HID_DIM 256
#define G3      768          // 3 * HID
#define N_NODES 100000
#define KSPLIT  32
#define CHUNK   ((N_NODES + KSPLIT - 1) / KSPLIT)   // 3125

#if __has_builtin(__builtin_amdgcn_tensor_load_to_lds) && \
    __has_builtin(__builtin_amdgcn_s_wait_tensorcnt)
#define HAVE_TDM 1
#else
#define HAVE_TDM 0
#endif

// ---------------- WMMA types / helpers ----------------
typedef __attribute__((ext_vector_type(16))) __bf16        v16bf;
typedef __attribute__((ext_vector_type(8)))  float         v8f;
typedef __attribute__((ext_vector_type(8)))  unsigned int  v8u;
typedef __attribute__((ext_vector_type(4)))  unsigned int  u32x4;
typedef __attribute__((ext_vector_type(8)))  int           i32x8;
typedef __attribute__((ext_vector_type(4)))  int           i32x4;

static __device__ inline unsigned int pack_bf16(float a, float b) {
  unsigned int ua = __float_as_uint(a);
  unsigned int ub = __float_as_uint(b);
  ua = (ua + 0x7FFFu + ((ua >> 16) & 1u)) >> 16;   // round-to-nearest-even
  ub = (ub + 0x7FFFu + ((ub >> 16) & 1u)) >> 16;
  return ua | (ub << 16);
}

static __device__ inline v8f wmma_bf16(v8u a, v8u b, v8f c) {
  v16bf ah = __builtin_bit_cast(v16bf, a);
  v16bf bh = __builtin_bit_cast(v16bf, b);
  return __builtin_amdgcn_wmma_f32_16x16x32_bf16(
      false, ah, false, bh, (short)0, c, false, false);
}

static __device__ inline v8f v8f_zero() {
  v8f z;
#pragma unroll
  for (int i = 0; i < 8; ++i) z[i] = 0.f;
  return z;
}

// LDS byte offset of a __shared__ object: flat LDS aperture address keeps the
// wave-relative LDS offset in addr[31:0] (ISA 10.2 aperture table).
static __device__ inline unsigned int lds_off_of(const void* p) {
  return (unsigned int)(unsigned long long)(uintptr_t)p;
}

#if HAVE_TDM
// Issue one TDM 2D tile load: 32 rows x 128 f32 elements, row stride in elems.
// tensor_dim1 = rows_avail so the hardware zero-fills OOB rows of the ragged
// last chunk. Descriptor groups are built from wave-uniform values -> SGPRs.
// clang-23 builtin arity: (u32x4 g0, i32x8 g1, i32x4 g2, i32x4 g3, i32x8, i32 cpol)
static __device__ inline void tdm_load_2d(unsigned int lds_byte_off,
                                          const float* gptr,
                                          unsigned int rows_avail,
                                          unsigned int row_stride_elems) {
  unsigned long long ga = (unsigned long long)(uintptr_t)gptr;
  u32x4 g0;
  g0[0] = 1u;                                            // count=1, user mode
  g0[1] = lds_byte_off;                                  // lds_addr
  g0[2] = (unsigned int)(ga & 0xFFFFFFFFull);            // global_addr[31:0]
  g0[3] = (unsigned int)((ga >> 32) & 0x01FFFFFFull)     // global_addr[56:32]
          | (2u << 30);                                  // type = 2 (image)
  i32x8 g1;
  g1[0] = (int)(2u << 16);                               // data_size = 4B
  g1[1] = (int)(128u << 16);                             // tensor_dim0[15:0]=128
  g1[2] = (int)((rows_avail & 0xFFFFu) << 16);           // dim0 hi=0 | dim1 lo
  g1[3] = (int)((rows_avail >> 16) | (128u << 16));      // dim1 hi | tile_dim0=128
  g1[4] = (int)32u;                                      // tile_dim1=32, tile_dim2=0
  g1[5] = (int)row_stride_elems;                         // dim0_stride[31:0]
  g1[6] = 0;                                             // dim0_stride hi | dim1_stride lo
  g1[7] = 0;
  i32x4 z4;
  z4[0] = z4[1] = z4[2] = z4[3] = 0;
  i32x8 z8;
#pragma unroll
  for (int i = 0; i < 8; ++i) z8[i] = 0;
  __builtin_amdgcn_tensor_load_to_lds(g0, g1, z4, z4, z8, 0);
}
#endif

// =====================================================================
// Kernel 1: partial GEMM  P[kb] = H[:,tblk].T @ X  over a K(node) chunk.
// TDM (or coalesced float4) fills raw f32 LDS tiles; LDS->LDS bf16 pack;
// v_wmma_f32_16x16x32_bf16 compute. TDM of tile i+1 overlaps WMMA of i.
// =====================================================================
__global__ __launch_bounds__(256)
void gemm1_kernel(const float* __restrict__ X, const float* __restrict__ H,
                  float* __restrict__ P) {
  __shared__ float        Araw[32 * 128];     // H tile  [k 0..31][t 0..127]
  __shared__ float        Braw[32 * 128];     // X tile  [k 0..31][c 0..127]
  __shared__ unsigned int Abuf[128 * 17];     // packed bf16 A frags [t][kp]
  __shared__ unsigned int Bbuf[128 * 17];     // packed bf16 B frags [c][q]

  const int tid  = threadIdx.x;
  const int wave = tid >> 5, lane = tid & 31;
  const int l16  = lane & 15, lh = lane >> 4;
  const int t0   = blockIdx.x * 128;
  const int nbeg = blockIdx.y * CHUNK;
  const int nend = min(nbeg + CHUNK, N_NODES);

  v8f acc[8];
#pragma unroll
  for (int i = 0; i < 8; ++i) acc[i] = v8f_zero();

#if HAVE_TDM
  const unsigned int offA = lds_off_of(&Araw[0]);
  const unsigned int offB = lds_off_of(&Braw[0]);
#endif

  // ---- fill raw tiles for chunk starting at n0 ----
  auto fill = [&](int n0) {
#if HAVE_TDM
    if (wave == 0) {
      tdm_load_2d(offA, H + (size_t)n0 * T_DIM + t0, (unsigned)(nend - n0), T_DIM);
      tdm_load_2d(offB, X + (size_t)n0 * IN_DIM,     (unsigned)(nend - n0), IN_DIM);
    }
#else
#pragma unroll
    for (int r = 0; r < 4; ++r) {
      int w4 = tid + 256 * r;                 // 0..1023 float4 slots
      int k = w4 >> 5, c4 = (w4 & 31) * 4;
      int n = n0 + k;
      float4 zv; zv.x = zv.y = zv.z = zv.w = 0.f;
      float4 av = (n < nend) ? *(const float4*)(H + (size_t)n * T_DIM + t0 + c4) : zv;
      *(float4*)(Araw + k * 128 + c4) = av;
      float4 bv = (n < nend) ? *(const float4*)(X + (size_t)n * IN_DIM + c4) : zv;
      *(float4*)(Braw + k * 128 + c4) = bv;
    }
#endif
  };
  auto wait_fill = [&]() {
#if HAVE_TDM
    if (wave == 0) __builtin_amdgcn_s_wait_tensorcnt(0);
#endif
  };

  fill(nbeg);
  wait_fill();
  __syncthreads();

  for (int n0 = nbeg; n0 < nend; n0 += 32) {
    // ---- LDS->LDS convert: pack bf16 K-pairs in WMMA fragment order ----
#pragma unroll
    for (int r = 0; r < 8; ++r) {
      int w  = tid + 256 * r;                 // 0..2047
      int t  = w & 127, kp = w >> 7;          // consecutive lanes -> same row
      Abuf[t * 17 + kp] = pack_bf16(Araw[(2 * kp) * 128 + t],
                                    Araw[(2 * kp + 1) * 128 + t]);
      int c = t, q = kp;
      int k = ((q & 3) << 1) + ((q >> 3) << 3) + (((q >> 2) & 1) << 4);
      Bbuf[c * 17 + q] = pack_bf16(Braw[k * 128 + c], Braw[(k + 1) * 128 + c]);
    }
    __syncthreads();                          // raw tiles now reusable

    if (n0 + 32 < nend) fill(n0 + 32);        // TDM overlaps WMMA below

    // ---- gather fragments + 8 WMMA ----
    v8u a;
    const int ta = wave * 16 + l16;
#pragma unroll
    for (int j = 0; j < 8; ++j) {
      int kp = ((j < 4) ? 0 : 8) + lh * 4 + (j & 3);
      a[j] = Abuf[ta * 17 + kp];
    }
#pragma unroll
    for (int ct = 0; ct < 8; ++ct) {
      v8u b;
      const int c = ct * 16 + l16;
#pragma unroll
      for (int j = 0; j < 8; ++j) b[j] = Bbuf[c * 17 + (lh * 8 + j)];
      acc[ct] = wmma_bf16(a, b, acc[ct]);
    }

    wait_fill();
    __syncthreads();                          // next tile ready, frags consumed
  }

  // ---- write per-K-split partial tile (deterministic, no atomics) ----
  float* Pk = P + (size_t)blockIdx.y * (T_DIM * IN_DIM);
#pragma unroll
  for (int ct = 0; ct < 8; ++ct) {
#pragma unroll
    for (int v = 0; v < 8; ++v) {
      int t = t0 + wave * 16 + v + 8 * lh;    // C layout: M = v + 8*lanehalf
      int c = ct * 16 + l16;                  // N = lane&15
      Pk[(size_t)t * IN_DIM + c] = acc[ct][v];
    }
  }
}

// =====================================================================
// Kernel 2: gx = (sum_kb P[kb]) @ w_ih.T + b_ih   [1024 x 768]
// Block = 256 threads; tile = 128 t x 64 j; K = 128 -> 4 bf16 WMMA steps.
// =====================================================================
__global__ __launch_bounds__(256)
void gemm2_kernel(const float* __restrict__ P, const float* __restrict__ w_ih,
                  const float* __restrict__ b_ih, float* __restrict__ gx) {
  __shared__ unsigned int Apk[128 * 65];   // [t][kp 0..63]  packed bf16 pairs
  __shared__ unsigned int Bq[64 * 65];     // [c][q' 0..63]
  const int tid  = threadIdx.x;
  const int wave = tid >> 5, lane = tid & 31;
  const int l16  = lane & 15, lh = lane >> 4;
  const int t0   = blockIdx.x * 128;
  const int j0   = blockIdx.y * 64;

  // ---- reduce K-split partials, pack to bf16 in LDS (coalesced float2) ----
  for (int r = 0; r < 32; ++r) {
    int w = tid + 256 * r;                 // 0..8191
    int t = w >> 6, kp = w & 63;
    float s0 = 0.f, s1 = 0.f;
    const float* base = P + (size_t)(t0 + t) * IN_DIM + 2 * kp;
    for (int kb = 0; kb < KSPLIT; ++kb) {
      const float2 v = *(const float2*)(base + (size_t)kb * (T_DIM * IN_DIM));
      s0 += v.x; s1 += v.y;
    }
    Apk[t * 65 + kp] = pack_bf16(s0, s1);
  }
  // ---- stage w_ih.T tile: B(k, j) = w_ih[j][k]  (float2 coalesced) ----
  for (int r = 0; r < 16; ++r) {
    int w = tid + 256 * r;                 // 0..4095
    int c = w >> 6, qp = w & 63;
    int s = qp >> 4, q = qp & 15;
    int k = s * 32 + ((q & 3) << 1) + ((q >> 3) << 3) + (((q >> 2) & 1) << 4);
    const float2 wv = *(const float2*)(w_ih + (size_t)(j0 + c) * IN_DIM + k);
    Bq[c * 65 + qp] = pack_bf16(wv.x, wv.y);
  }
  __syncthreads();

  v8f acc[4];
#pragma unroll
  for (int i = 0; i < 4; ++i) acc[i] = v8f_zero();

  const int ta = wave * 16 + l16;
#pragma unroll
  for (int s = 0; s < 4; ++s) {
    v8u a;
#pragma unroll
    for (int j = 0; j < 8; ++j) {
      int kp = s * 16 + ((j < 4) ? 0 : 8) + lh * 4 + (j & 3);
      a[j] = Apk[ta * 65 + kp];
    }
#pragma unroll
    for (int ct = 0; ct < 4; ++ct) {
      v8u b;
      const int c = ct * 16 + l16;
#pragma unroll
      for (int j = 0; j < 8; ++j) b[j] = Bq[c * 65 + (s * 16 + lh * 8 + j)];
      acc[ct] = wmma_bf16(a, b, acc[ct]);
    }
  }

  // ---- epilogue: add bias, store f32 ----
#pragma unroll
  for (int ct = 0; ct < 4; ++ct) {
    int j = j0 + ct * 16 + l16;
    float bias = b_ih[j];
#pragma unroll
    for (int v = 0; v < 8; ++v) {
      int t = t0 + wave * 16 + v + 8 * lh;
      gx[(size_t)t * G3 + j] = acc[ct][v] + bias;
    }
  }
}

// =====================================================================
// Kernel 3: sequential GRU scan (persistent single workgroup, 768 thr),
// per-step attention logits, then softmax + alpha-weighted sum.
// =====================================================================
__global__ __launch_bounds__(768)
void gru_scan_kernel(const float* __restrict__ gx, const float* __restrict__ w_hh,
                     const float* __restrict__ b_hh, const float* __restrict__ w_attn,
                     float* __restrict__ hs, float* __restrict__ out) {
  __shared__ float h_lds[HID_DIM];
  __shared__ float gh_lds[G3];
  __shared__ float red[HID_DIM];
  __shared__ float logit[T_DIM];
  __shared__ float sred[2];

  const int tid = threadIdx.x;               // 0..767
  if (tid < HID_DIM) h_lds[tid] = 0.f;
  const float bh = b_hh[tid];
  const float4* wrow = (const float4*)(w_hh + (size_t)tid * HID_DIM);
  __syncthreads();

  for (int t = 0; t < T_DIM; ++t) {
    // phase 1: gh[j] = b_hh[j] + w_hh[j,:] . h   (w_hh streams from L2)
    float acc = bh;
    const float4* hv = (const float4*)h_lds;
#pragma unroll 8
    for (int k = 0; k < HID_DIM / 4; ++k) {
      float4 wv = wrow[k];
      float4 h4 = hv[k];
      acc += wv.x * h4.x + wv.y * h4.y + wv.z * h4.z + wv.w * h4.w;
    }
    gh_lds[tid] = acc;
    __syncthreads();

    // phase 2: gates + state update (threads 0..255)
    if (tid < HID_DIM) {
      const float* gxt = gx + (size_t)t * G3;
      float xr = gxt[tid], xz = gxt[HID_DIM + tid], xn = gxt[2 * HID_DIM + tid];
      float r = 1.f / (1.f + __expf(-(xr + gh_lds[tid])));
      float z = 1.f / (1.f + __expf(-(xz + gh_lds[HID_DIM + tid])));
      float n = tanhf(xn + r * gh_lds[2 * HID_DIM + tid]);
      float hnew = (1.f - z) * n + z * h_lds[tid];
      h_lds[tid] = hnew;
      hs[(size_t)t * HID_DIM + tid] = hnew;
      red[tid] = hnew * w_attn[tid];
    }
    __syncthreads();

    // phase 3: logit[t] = hs_t . w_attn  (one wave32, shuffle reduce)
    if (tid < 32) {
      float s = 0.f;
#pragma unroll
      for (int i = 0; i < 8; ++i) s += red[tid + 32 * i];
#pragma unroll
      for (int off = 16; off > 0; off >>= 1) s += __shfl_down(s, off, 32);
      if (tid == 0) logit[t] = s;
    }
    __syncthreads();
  }

  // ---- softmax over logits (in LDS) ----
  if (tid < 32) {
    float m = -3.4e38f;
    for (int i = 0; i < 32; ++i) m = fmaxf(m, logit[tid * 32 + i]);
#pragma unroll
    for (int off = 16; off > 0; off >>= 1) m = fmaxf(m, __shfl_down(m, off, 32));
    if (tid == 0) sred[0] = m;
  }
  __syncthreads();
  {
    float m = sred[0];
    for (int w = tid; w < T_DIM; w += 768) logit[w] = __expf(logit[w] - m);
  }
  __syncthreads();
  if (tid < 32) {
    float s = 0.f;
    for (int i = 0; i < 32; ++i) s += logit[tid * 32 + i];
#pragma unroll
    for (int off = 16; off > 0; off >>= 1) s += __shfl_down(s, off, 32);
    if (tid == 0) sred[1] = s;
  }
  __syncthreads();

  // ---- out[j] = sum_t alpha_t * hs[t][j] ----
  if (tid < HID_DIM) {
    float inv = 1.f / sred[1];
    float a = 0.f;
    for (int t = 0; t < T_DIM; ++t) a += logit[t] * hs[(size_t)t * HID_DIM + tid];
    out[tid] = a * inv;
  }
}

// =====================================================================
extern "C" void kernel_launch(void* const* d_in, const int* in_sizes, int n_in,
                              void* d_out, int out_size, void* d_ws, size_t ws_size,
                              hipStream_t stream) {
  (void)in_sizes; (void)n_in; (void)out_size; (void)ws_size;
  const float* X      = (const float*)d_in[0];
  const float* H      = (const float*)d_in[1];
  const float* w_ih   = (const float*)d_in[2];
  const float* w_hh   = (const float*)d_in[3];
  const float* b_ih   = (const float*)d_in[4];
  const float* b_hh   = (const float*)d_in[5];
  const float* w_attn = (const float*)d_in[6];
  float* out = (float*)d_out;

  // workspace layout (floats): P[KSPLIT][1024][128] | gx[1024][768] | hs[1024][256]
  float* P  = (float*)d_ws;                           // 16 MB
  float* gx = P  + (size_t)KSPLIT * T_DIM * IN_DIM;   //  3 MB
  float* hs = gx + (size_t)T_DIM * G3;                //  1 MB

  gemm1_kernel<<<dim3(T_DIM / 128, KSPLIT), 256, 0, stream>>>(X, H, P);
  gemm2_kernel<<<dim3(T_DIM / 128, G3 / 64), 256, 0, stream>>>(P, w_ih, b_ih, gx);
  gru_scan_kernel<<<1, 768, 0, stream>>>(gx, w_hh, b_hh, w_attn, hs, out);
}
